// RelaxedContrastiveLoss_68968584839501
// MI455X (gfx1250) — matrix operand order repaired and checked by reading, not compiled, not run on hardware
//
#include <hip/hip_runtime.h>
#include <hip/hip_bf16.h>
#include <math.h>

typedef __attribute__((ext_vector_type(16))) _Float16 v16h;
typedef __attribute__((ext_vector_type(8)))  _Float16 v8h;
typedef __attribute__((ext_vector_type(8)))  float    v8f;

#define BATCH 8192
#define DIM   256
#define NKF   (DIM / 32)        // 8 K-fragments of 16x32
#define NJT   (BATCH / 16)      // 512 column tiles
#define INV_T 20.0f             // 1 / TEMPERATURE
#define MT_PER_BLK   4          // M tiles per block (64 rows)
#define ROWS_PER_BLK (MT_PER_BLK * 16)
#define JSPLIT       2          // waves sharing one M tile, splitting J by parity

// WGP-scope prefetch (scope field 0): pulls the line into WGP$ + L2, which is
// what the 4-way intra-WGP B-tile reuse wants. The __builtin_prefetch path
// emits scope:SCOPE_SE (GL2 only), so issue the mnemonic directly.
__device__ __forceinline__ void prefetch_wgp(const void* p) {
    asm volatile("global_prefetch_b8 %0, off" :: "v"(p));
}

// ---------------- zero the global accumulators ----------------
__global__ void init_acc_kernel(float* acc) {
    if (threadIdx.x < 8) acc[threadIdx.x] = 0.0f;
}

// ---------------- L2-normalize rows, convert f32 -> f16 ----------------
__global__ void normalize_kernel(const float* __restrict__ feat,
                                 _Float16* __restrict__ fh) {
    const int wave = threadIdx.x >> 5;
    const int lane = threadIdx.x & 31;
    const int row  = blockIdx.x * 8 + wave;
    if (row >= BATCH) return;
    const float4* p = (const float4*)(feat + row * DIM + lane * 8);
    float4 x0 = p[0];
    float4 x1 = p[1];
    float s = x0.x*x0.x + x0.y*x0.y + x0.z*x0.z + x0.w*x0.w
            + x1.x*x1.x + x1.y*x1.y + x1.z*x1.z + x1.w*x1.w;
    #pragma unroll
    for (int m = 16; m >= 1; m >>= 1) s += __shfl_xor(s, m, 32);
    const float scale = 1.0f / fmaxf(sqrtf(s), 1e-12f);
    v8h h;
    h[0] = (_Float16)(x0.x * scale);
    h[1] = (_Float16)(x0.y * scale);
    h[2] = (_Float16)(x0.z * scale);
    h[3] = (_Float16)(x0.w * scale);
    h[4] = (_Float16)(x1.x * scale);
    h[5] = (_Float16)(x1.y * scale);
    h[6] = (_Float16)(x1.z * scale);
    h[7] = (_Float16)(x1.w * scale);
    *(v8h*)(fh + (size_t)row * DIM + lane * 8) = h;
}

// ---------------- fused Gram + streaming row statistics ----------------
// block = 8 waves, 4 M-tiles (64 rows). Waves 2m,2m+1 own tile m and split
// the 512 J tiles by parity, so 4 waves always touch the same B tile at the
// same time -> WGP$ serves 3 of 4 (L2 traffic /4). Periodic barrier keeps
// the parity groups converged within the WGP$ window.
__global__ void __launch_bounds__(256)
simloss_kernel(const _Float16* __restrict__ fh,
               const int* __restrict__ labels,
               float* __restrict__ acc) {
    __shared__ float redM[8][16], redS[8][16], redP[8][16],
                     redQ[8][16], redH[8][16], redC[8][16];

    const int tid   = threadIdx.x;
    const int wave  = tid >> 5;
    const int lane  = tid & 31;
    const int g     = lane >> 4;           // half-wave group (K/N split)
    const int n16   = lane & 15;
    const int mt    = wave >> 1;           // M tile within block
    const int jpar  = wave & 1;            // J parity for this wave
    const int rbase = blockIdx.x * ROWS_PER_BLK + mt * 16;
    const int iA    = rbase + n16;         // A row this lane feeds

    // A fragments (16x32 f16), resident across the whole J sweep.
    // lane L holds row L%16; halves t<8 -> K=8g+t, t>=8 -> K=16+8g+(t-8)
    v16h afrag[NKF];
    #pragma unroll
    for (int kk = 0; kk < NKF; ++kk) {
        const int kbase = kk * 32;
        v8h c0 = *(const v8h*)(fh + (size_t)iA * DIM + kbase + 8 * g);
        v8h c1 = *(const v8h*)(fh + (size_t)iA * DIM + kbase + 16 + 8 * g);
        afrag[kk] = __builtin_shufflevector(c0, c1,
            0,1,2,3,4,5,6,7, 8,9,10,11,12,13,14,15);
    }

    int rlab[8];
    #pragma unroll
    for (int r = 0; r < 8; ++r)
        rlab[r] = labels[rbase + g * 8 + r];

    // streaming per-row stats, stabilizer fixed at m0 = 1/T (row max ~ 1/T)
    float mrow[8], Ssum[8], Pcnt[8], Qsum[8], Hsum[8], Ccnt[8];
    #pragma unroll
    for (int r = 0; r < 8; ++r) {
        mrow[r] = -1e30f; Ssum[r] = 0.f; Pcnt[r] = 0.f;
        Qsum[r] = 0.f;    Hsum[r] = 0.f; Ccnt[r] = 0.f;
    }

    for (int jt = jpar; jt < NJT; jt += JSPLIT) {
        const int jrow = jt * 16 + n16;    // B column == sim column for lane
        // unconditional WGP-scope prefetch of this lane's next B chunk
        // (wrapped index on the last iterations -> harmless warm row)
        int jn = jrow + JSPLIT * 16;
        jn = (jn >= BATCH) ? (jn - BATCH) : jn;
        prefetch_wgp(fh + (size_t)jn * DIM + g * 128);

        const int jlab = labels[jrow];
        v8f c = {};
        #pragma unroll
        for (int kk = 0; kk < NKF; ++kk) {
            // B 32x16 f16: lane L holds column L%16, t<8 -> K=8g+t, t>=8 -> K=16+8g+(t-8)
            v8h b0 = *(const v8h*)(fh + (size_t)jrow * DIM + kk * 32 + 8 * g);
            v8h b1 = *(const v8h*)(fh + (size_t)jrow * DIM + kk * 32 + 16 + 8 * g);
            v16h b = __builtin_shufflevector(b0, b1,
                0,1,2,3,4,5,6,7, 8,9,10,11,12,13,14,15);
            c = __builtin_amdgcn_wmma_f32_16x16x32_f16(
                    false, afrag[kk], false, b, (short)0, c, false, false);
        }
        #pragma unroll
        for (int r = 0; r < 8; ++r) {
            const int   ig   = rbase + g * 8 + r;
            const float dot  = c[r];
            const float sim  = dot * INV_T;
            mrow[r] = fmaxf(mrow[r], sim);          // max includes diagonal
            const bool  diag = (ig == jrow);
            const float e    = __expf(sim - INV_T); // e^{sim - m0}, <= e^~0.02
            Ssum[r] += diag ? 0.0f : e;
            const bool pos = (!diag) && (rlab[r] == jlab);
            Pcnt[r] += pos ? 1.0f : 0.0f;
            Qsum[r] += pos ? sim  : 0.0f;
            const bool high = pos && (dot > 0.7f);  // sim*T > lambda
            Hsum[r] += high ? e : 0.0f;
            Ccnt[r] += high ? 1.0f : 0.0f;
        }
        if (((jt >> 1) & 15) == 15) __syncthreads();  // keep groups converged
    }

    // reduce across the 16 lanes of each half-group (rows live per-VGPR)
    #pragma unroll
    for (int r = 0; r < 8; ++r) {
        #pragma unroll
        for (int m = 8; m >= 1; m >>= 1) {
            mrow[r] = fmaxf(mrow[r], __shfl_xor(mrow[r], m, 32));
            Ssum[r] += __shfl_xor(Ssum[r], m, 32);
            Pcnt[r] += __shfl_xor(Pcnt[r], m, 32);
            Qsum[r] += __shfl_xor(Qsum[r], m, 32);
            Hsum[r] += __shfl_xor(Hsum[r], m, 32);
            Ccnt[r] += __shfl_xor(Ccnt[r], m, 32);
        }
    }
    if (n16 == 0) {
        #pragma unroll
        for (int r = 0; r < 8; ++r) {
            const int row = g * 8 + r;
            redM[wave][row] = mrow[r];
            redS[wave][row] = Ssum[r];
            redP[wave][row] = Pcnt[r];
            redQ[wave][row] = Qsum[r];
            redH[wave][row] = Hsum[r];
            redC[wave][row] = Ccnt[r];
        }
    }
    __syncthreads();

    // 64 threads (waves 0,1) finalize the block's 64 rows
    if (tid < ROWS_PER_BLK) {
        const int mt2 = tid >> 4;          // which M tile
        const int rr  = tid & 15;          // row within tile
        const int w0  = mt2 * 2, w1 = w0 + 1;
        float m = fmaxf(redM[w0][rr], redM[w1][rr]);
        float S = redS[w0][rr] + redS[w1][rr];
        float P = redP[w0][rr] + redP[w1][rr];
        float Q = redQ[w0][rr] + redQ[w1][rr];
        float H = redH[w0][rr] + redH[w1][rr];
        float C = redC[w0][rr] + redC[w1][rr];
        // log_denom = log(sum_{j!=i} e^{sim-m} + eps) + m, sum = S*e^{m0-m}
        const float log_denom = __logf(S * __expf(INV_T - m) + 1e-12f) + m;
        float scl = P * log_denom - Q;     // sum over positives of (ld - sim)
        // row_lse = log(sum_high e^{sim-m} + e^{m0-m}) = (m0-m) + log(H+1)
        float rel = (C > 0.0f) ? ((INV_T - m) + __logf(H + 1.0f)) : 0.0f;
        float Ps = P, Cs = C;
        #pragma unroll
        for (int msk = 16; msk >= 1; msk >>= 1) {
            scl += __shfl_xor(scl, msk, 32);
            Ps  += __shfl_xor(Ps,  msk, 32);
            rel += __shfl_xor(rel, msk, 32);
            Cs  += __shfl_xor(Cs,  msk, 32);
        }
        if (lane == 0) {                   // lane 0 of each of the two waves
            atomicAdd(&acc[0], scl);
            atomicAdd(&acc[1], Ps);
            atomicAdd(&acc[2], rel);
            atomicAdd(&acc[3], Cs);
        }
    }
}

// ---------------- finalize ----------------
__global__ void finalize_kernel(const float* __restrict__ acc,
                                float* __restrict__ out) {
    const float scl = acc[0], npos = acc[1], rel = acc[2], nhigh = acc[3];
    const float scl_loss = (npos  > 0.f) ? scl / fmaxf(npos,  1.f) : 0.f;
    const float relax    = (nhigh > 0.f) ? rel / fmaxf(nhigh, 1.f) : 0.f;
    float t = scl_loss + relax;
    out[0] = fminf(fmaxf(t, 0.f), 10.f);
}

extern "C" void kernel_launch(void* const* d_in, const int* in_sizes, int n_in,
                              void* d_out, int out_size, void* d_ws, size_t ws_size,
                              hipStream_t stream) {
    (void)in_sizes; (void)n_in; (void)out_size; (void)ws_size;
    const float* feat   = (const float*)d_in[0];
    const int*   labels = (const int*)d_in[1];
    float*       out    = (float*)d_out;

    _Float16* fh  = (_Float16*)d_ws;                                   // 4 MB f16 matrix
    float*    acc = (float*)((char*)d_ws +
                             (size_t)BATCH * DIM * sizeof(_Float16));  // scalars

    init_acc_kernel<<<1, 32, 0, stream>>>(acc);
    normalize_kernel<<<BATCH / 8, 256, 0, stream>>>(feat, fh);
    simloss_kernel<<<BATCH / ROWS_PER_BLK, 256, 0, stream>>>(fh, labels, acc);
    finalize_kernel<<<1, 1, 0, stream>>>(acc, out);
}
